// MultiHeadAttention_86998857548098
// MI455X (gfx1250) — matrix operand order repaired and checked by reading, not compiled, not run on hardware
//
#include <hip/hip_runtime.h>
#include <stdint.h>

#define B_  2
#define S_  2048
#define D_  1024
#define H_  16
#define DH_ 64

typedef _Float16 h8_t  __attribute__((ext_vector_type(8)));
typedef _Float16 h16_t __attribute__((ext_vector_type(16)));
typedef float    f8_t  __attribute__((ext_vector_type(8)));

static __device__ __forceinline__ h16_t cat16(h8_t lo, h8_t hi) {
  return __builtin_shufflevector(lo, hi, 0,1,2,3,4,5,6,7,8,9,10,11,12,13,14,15);
}

static __device__ __forceinline__ f8_t wmma_f16(h16_t a, h16_t b, f8_t c) {
  // D = A(16x32 f16) * B(32x16 f16) + C(16x16 f32)
  return __builtin_amdgcn_wmma_f32_16x16x32_f16(false, a, false, b, (short)0, c, false, false);
}

// A fragment (16x32 f16). base -> tile (row0,k0), stride in halfs.
static __device__ __forceinline__ h16_t ldsA(const _Float16* base, int stride, int lane) {
  const int r = lane & 15, hf = lane >> 4;
  const _Float16* p = base + r * stride + hf * 8;
  h8_t lo = *(const h8_t*)p;         // K = hf*8 .. +7
  h8_t hi = *(const h8_t*)(p + 16);  // K = 16+hf*8 .. +7
  return cat16(lo, hi);
}

// B fragment (32x16 f16) from LDS tile stored [n][k].
static __device__ __forceinline__ h16_t ldsB(const _Float16* base, int stride, int lane) {
  const int n = lane & 15, kh = lane >> 4;
  const _Float16* p = base + n * stride + kh * 8;
  h8_t lo = *(const h8_t*)p;         // K = kh*8 .. +7
  h8_t hi = *(const h8_t*)(p + 16);  // K = 16+kh*8 .. +7
  return cat16(lo, hi);
}

static __device__ __forceinline__ h8_t cvt8(float4 a, float4 b) {
  h8_t h;
  h[0]=(_Float16)a.x; h[1]=(_Float16)a.y; h[2]=(_Float16)a.z; h[3]=(_Float16)a.w;
  h[4]=(_Float16)b.x; h[5]=(_Float16)b.y; h[6]=(_Float16)b.z; h[7]=(_Float16)b.w;
  return h;
}

// CDNA5 async LDS copy: lds[VDST + lane] <- mem[VADDR(lane)], tracked on ASYNCcnt.
static __device__ __forceinline__ void async_b128(unsigned lds_byte_off, const void* gptr) {
  unsigned long long ga = (unsigned long long)(uintptr_t)gptr;
  asm volatile("global_load_async_to_lds_b128 %0, %1, off"
               :: "v"(lds_byte_off), "v"(ga) : "memory");
}
#define WAIT_ASYNCCNT(imm) asm volatile("s_wait_asynccnt " imm ::: "memory")

// ---------------------------------------------------------------------------
// Kernel 1: fused cast + projection GEMM.  C[4096x1024] = X @ W + bias.
// z=0: Q -> [B,H,S,DH] f16 ; z=1: K -> [B,H,S,DH] f16 ; z=2: V -> [B,H,DH,S] f16
// ---------------------------------------------------------------------------
#define PTM 128
#define PTN 128
#define PTK 32
#define ASTR 40
#define BSTR 40

__global__ __launch_bounds__(256)
void proj_qkv(const float* __restrict__ Xq, const float* __restrict__ Xk, const float* __restrict__ Xv,
              const float* __restrict__ Wq, const float* __restrict__ Wk, const float* __restrict__ Wv,
              const float* __restrict__ bq, const float* __restrict__ bk, const float* __restrict__ bv,
              _Float16* __restrict__ Qo, _Float16* __restrict__ Ko, _Float16* __restrict__ Vo)
{
  __shared__ _Float16 As[PTM * ASTR];   // X tile, row-major [m][k]
  __shared__ _Float16 Bs[PTN * BSTR];   // W tile, transposed  [n][k]

  const int z = blockIdx.z;
  const float* X    = (z == 0) ? Xq : (z == 1) ? Xk : Xv;
  const float* W    = (z == 0) ? Wq : (z == 1) ? Wk : Wv;
  const float* bias = (z == 0) ? bq : (z == 1) ? bk : bv;
  _Float16*    O    = (z == 0) ? Qo : (z == 1) ? Ko : Vo;
  const bool vtrans = (z == 2);

  const int t    = threadIdx.x;
  const int lane = t & 31;
  const int wave = t >> 5;
  const int wm   = wave & 3;
  const int wn   = wave >> 2;
  const int gm0  = blockIdx.y * PTM;
  const int gn0  = blockIdx.x * PTN;

  f8_t acc[2][4];
#pragma unroll
  for (int i = 0; i < 2; ++i)
#pragma unroll
    for (int j = 0; j < 4; ++j)
      acc[i][j] = (f8_t){0.f,0.f,0.f,0.f,0.f,0.f,0.f,0.f};

  for (int kk = 0; kk < D_; kk += PTK) {
    __syncthreads();
    {
      const int row = t >> 1, c0 = (t & 1) * 16;
      const float4* src = (const float4*)(X + (size_t)(gm0 + row) * D_ + kk + c0);
      float4 v0 = src[0], v1 = src[1], v2 = src[2], v3 = src[3];
      *(h8_t*)&As[row * ASTR + c0]     = cvt8(v0, v1);
      *(h8_t*)&As[row * ASTR + c0 + 8] = cvt8(v2, v3);
    }
    {
      const int k = t >> 3, n0 = (t & 7) * 16;
      const float4* src = (const float4*)(W + (size_t)(kk + k) * D_ + gn0 + n0);
      float4 w0 = src[0], w1 = src[1], w2 = src[2], w3 = src[3];
      float f[16] = {w0.x,w0.y,w0.z,w0.w, w1.x,w1.y,w1.z,w1.w,
                     w2.x,w2.y,w2.z,w2.w, w3.x,w3.y,w3.z,w3.w};
#pragma unroll
      for (int j = 0; j < 16; ++j)
        Bs[(n0 + j) * BSTR + k] = (_Float16)f[j];
    }
    __syncthreads();

    h16_t af0 = ldsA(&As[(wm * 32 +  0) * ASTR], ASTR, lane);
    h16_t af1 = ldsA(&As[(wm * 32 + 16) * ASTR], ASTR, lane);
#pragma unroll
    for (int tj = 0; tj < 4; ++tj) {
      h16_t bf = ldsB(&Bs[(wn * 64 + tj * 16) * BSTR], BSTR, lane);
      acc[0][tj] = wmma_f16(af0, bf, acc[0][tj]);
      acc[1][tj] = wmma_f16(af1, bf, acc[1][tj]);
    }
  }

  const int cn = lane & 15, chalf = lane >> 4;
#pragma unroll
  for (int ti = 0; ti < 2; ++ti)
#pragma unroll
    for (int tj = 0; tj < 4; ++tj) {
      const int gn = gn0 + wn * 64 + tj * 16 + cn;
      const float bval = bias[gn];
      const int hh = gn >> 6, dd = gn & 63;
#pragma unroll
      for (int i = 0; i < 8; ++i) {
        const int gm = gm0 + wm * 32 + ti * 16 + i + chalf * 8;
        const int bb = gm >> 11, ss = gm & 2047;
        const float val = acc[ti][tj][i] + bval;
        const size_t idx = vtrans
            ? ((size_t)((bb * H_ + hh) * DH_ + dd)) * S_ + ss    // V^T: [B,H,DH,S]
            : ((size_t)((bb * H_ + hh) * S_ + ss)) * DH_ + dd;   // Q,K: [B,H,S,DH]
        O[idx] = (_Float16)val;
      }
    }
}

// ---------------------------------------------------------------------------
// Kernel 2: flash attention, async-LDS double-buffered K/V tiles.
// Block = (b,h) x 128 queries; 8 waves x 16 q rows; 16 key tiles of 128.
// ---------------------------------------------------------------------------
#define QT 128
#define KT 128
#define NKT (S_ / KT)   // 16
#define QSTR 72
#define KSTR 72
#define VSTR 136
#define PSTR 136

#define KS_HALFS (KT * KSTR)          // 9216
#define VS_HALFS (DH_ * VSTR)         // 8704
#define TILE_HALFS (KS_HALFS + VS_HALFS)
#define QP_HALFS (8 * 16 * PSTR)      // 17408 (>= 128*QSTR; Q tile then P buffers)
#define ATTN_LDS_BYTES ((2 * TILE_HALFS + QP_HALFS) * 2)   // 106496 B

__global__ __launch_bounds__(256)
void attn(const _Float16* __restrict__ Q, const _Float16* __restrict__ K,
          const _Float16* __restrict__ Vt, float* __restrict__ out)
{
  extern __shared__ _Float16 smem[];
  _Float16* qp = smem + 2 * TILE_HALFS;   // Q tile, later per-wave P buffers

  const int t = threadIdx.x, lane = t & 31, wave = t >> 5;
  const int bh = blockIdx.y, b = bh >> 4, h = bh & 15;
  const int q0 = blockIdx.x * QT;
  const size_t baseQK = ((size_t)(b * H_ + h) * S_) * DH_;   // [B,H,S,DH]
  const size_t baseV  = ((size_t)(b * H_ + h) * DH_) * S_;   // [B,H,DH,S]

  // per-thread staging coordinates (byte offsets within dynamic LDS)
  const int krow = t >> 1, kc0 = (t & 1) * 32;     // K/Q tile: 128 rows x 64 halfs
  const int vrow = t >> 2, vc0 = (t & 3) * 32;     // V^T tile:  64 rows x 128 halfs

  // issue the 8 async b128 loads (4 K + 4 V^T) for key-tile kt into buffer `buf`
  auto issue_tile = [&](int kt, int buf) {
    const unsigned kbase = (unsigned)(buf * TILE_HALFS * 2) + (unsigned)((krow * KSTR + kc0) * 2);
    const _Float16* kg = K + baseQK + (size_t)(kt * KT + krow) * DH_ + kc0;
    const unsigned vbase = (unsigned)(buf * TILE_HALFS * 2) + (unsigned)(KS_HALFS * 2)
                         + (unsigned)((vrow * VSTR + vc0) * 2);
    const _Float16* vg = Vt + baseV + (size_t)vrow * S_ + kt * KT + vc0;
#pragma unroll
    for (int j = 0; j < 4; ++j) async_b128(kbase + j * 16, (const char*)kg + j * 16);
#pragma unroll
    for (int j = 0; j < 4; ++j) async_b128(vbase + j * 16, (const char*)vg + j * 16);
  };

  // stage Q tile [128][QSTR] asynchronously, then prefetch key-tile 0
  {
    const unsigned qbase = (unsigned)(2 * TILE_HALFS * 2) + (unsigned)((krow * QSTR + kc0) * 2);
    const _Float16* qg = Q + baseQK + (size_t)(q0 + krow) * DH_ + kc0;
#pragma unroll
    for (int j = 0; j < 4; ++j) async_b128(qbase + j * 16, (const char*)qg + j * 16);
  }
  issue_tile(0, 0);
  WAIT_ASYNCCNT("0x8");   // async retires in order: Q done, tile0 (8) still in flight
  __syncthreads();

  // per-wave Q A-fragments, kept in registers for the whole kernel
  h16_t qa[2];
#pragma unroll
  for (int kc = 0; kc < 2; ++kc)
    qa[kc] = ldsA(&qp[(wave * 16) * QSTR + kc * 32], QSTR, lane);

  float mrow[8], lrow[8];
  f8_t  oacc[4];
#pragma unroll
  for (int i = 0; i < 8; ++i) { mrow[i] = -1e30f; lrow[i] = 0.f; }
#pragma unroll
  for (int dt = 0; dt < 4; ++dt)
    oacc[dt] = (f8_t){0.f,0.f,0.f,0.f,0.f,0.f,0.f,0.f};

  const float scale = 0.125f;            // 1/sqrt(DH)
  const int cn = lane & 15, chalf = lane >> 4;
  _Float16* pw = &qp[wave * 16 * PSTR];  // this wave's 16x128 P buffer

  for (int kt = 0; kt < NKT; ++kt) {
    __syncthreads();                     // all waves done with back buffer -> it is free
    if (kt + 1 < NKT) {
      issue_tile(kt + 1, (kt + 1) & 1);  // overlap next tile's DMA with this tile's math
      WAIT_ASYNCCNT("0x8");              // wait for tile kt (8 newest = tile kt+1 stay in flight)
    } else {
      WAIT_ASYNCCNT("0x0");
    }
    __syncthreads();                     // tile kt visible to all waves

    const _Float16* ksb = smem + (size_t)(kt & 1) * TILE_HALFS;
    const _Float16* vsb = ksb + KS_HALFS;

    // scores: 16 queries x 128 keys = 8 N-tiles, K-dim = DH (2 chunks of 32)
    f8_t sc[8];
#pragma unroll
    for (int tj = 0; tj < 8; ++tj) {
      f8_t zacc = (f8_t){0.f,0.f,0.f,0.f,0.f,0.f,0.f,0.f};
#pragma unroll
      for (int kc = 0; kc < 2; ++kc) {
        h16_t bf = ldsB(&ksb[(tj * 16) * KSTR + kc * 32], KSTR, lane);
        zacc = wmma_f16(qa[kc], bf, zacc);
      }
      sc[tj] = zacc;
    }

    // online softmax; row m = i + 8*chalf, col = tj*16 + cn; row-reduce via 16-lane xor shuffles
#pragma unroll
    for (int i = 0; i < 8; ++i) {
      float mx = -1e30f;
#pragma unroll
      for (int tj = 0; tj < 8; ++tj) {
        sc[tj][i] *= scale;
        mx = fmaxf(mx, sc[tj][i]);
      }
#pragma unroll
      for (int m = 1; m < 16; m <<= 1) mx = fmaxf(mx, __shfl_xor(mx, m, 32));
      const float mnew = fmaxf(mrow[i], mx);
      const float corr = __expf(mrow[i] - mnew);
      float rsum = 0.f;
#pragma unroll
      for (int tj = 0; tj < 8; ++tj) {
        const float p = __expf(sc[tj][i] - mnew);
        sc[tj][i] = p;
        rsum += p;
      }
#pragma unroll
      for (int m = 1; m < 16; m <<= 1) rsum += __shfl_xor(rsum, m, 32);
      lrow[i] = lrow[i] * corr + rsum;
      mrow[i] = mnew;
#pragma unroll
      for (int dt = 0; dt < 4; ++dt) oacc[dt][i] *= corr;
    }

    // C-layout P -> A-layout P via per-wave LDS buffer (same-wave DS ops are in-order)
#pragma unroll
    for (int tj = 0; tj < 8; ++tj)
#pragma unroll
      for (int i = 0; i < 8; ++i)
        pw[(i + chalf * 8) * PSTR + tj * 16 + cn] = (_Float16)sc[tj][i];

    h16_t pa[4];
#pragma unroll
    for (int kc = 0; kc < 4; ++kc)
      pa[kc] = ldsA(&pw[kc * 32], PSTR, lane);

    // O += P * V : 4 d-tiles x 4 key-chunks of 32
#pragma unroll
    for (int dt = 0; dt < 4; ++dt)
#pragma unroll
      for (int kc = 0; kc < 4; ++kc) {
        h16_t vb = ldsB(&vsb[(dt * 16) * VSTR + kc * 32], VSTR, lane);
        oacc[dt] = wmma_f16(pa[kc], vb, oacc[dt]);
      }
  }

  // epilogue: O / l, fp32 out at [b, s, h*64 + d]
#pragma unroll
  for (int i = 0; i < 8; ++i) {
    const float inv = 1.0f / lrow[i];
    const int qrow = q0 + wave * 16 + i + chalf * 8;
    float* orow = out + ((size_t)(b * S_ + qrow)) * D_ + h * DH_;
#pragma unroll
    for (int dt = 0; dt < 4; ++dt)
      orow[dt * 16 + cn] = oacc[dt][i] * inv;
  }
}

// ---------------------------------------------------------------------------
extern "C" void kernel_launch(void* const* d_in, const int* in_sizes, int n_in,
                              void* d_out, int out_size, void* d_ws, size_t ws_size,
                              hipStream_t stream) {
  (void)in_sizes; (void)n_in; (void)out_size; (void)ws_size;
  const float* v_q = (const float*)d_in[0];
  const float* v_k = (const float*)d_in[1];
  const float* v_v = (const float*)d_in[2];
  const float* wq  = (const float*)d_in[3];
  const float* bq  = (const float*)d_in[4];
  const float* wk  = (const float*)d_in[5];
  const float* bk  = (const float*)d_in[6];
  const float* wv  = (const float*)d_in[7];
  const float* bv  = (const float*)d_in[8];

  const size_t elems = (size_t)B_ * S_ * D_;
  _Float16* qws = (_Float16*)d_ws;
  _Float16* kws = qws + elems;
  _Float16* vws = kws + elems;   // 24 MB f16 workspace

  dim3 pgrid(D_ / PTN, (B_ * S_) / PTM, 3);
  proj_qkv<<<pgrid, 256, 0, stream>>>(v_q, v_k, v_v, wq, wk, wv, bq, bk, bv,
                                      qws, kws, vws);

  dim3 agrid(S_ / QT, B_ * H_);
  attn<<<agrid, 256, ATTN_LDS_BYTES, stream>>>(qws, kws, vws, (float*)d_out);
}
